// RQBottleneck_3728031613671
// MI455X (gfx1250) — matrix operand order repaired and checked by reading, not compiled, not run on hardware
//
#include <hip/hip_runtime.h>
#include <stdint.h>

// ---------------------------------------------------------------------------
// RQ bottleneck for MI455X (gfx1250, wave32, WMMA).
//   per level: fused [8192x256]x[256x8192] bf16-WMMA distance GEMM + argmin
//   epilogue (accumulator pre-loaded with -0.5*||c||^2, so argmin(dist) ==
//   argmax(accum)), then a finalize pass: gather + residual update + loss.
// Round 4: TDM staging via the 6-arg tensor_load_to_lds builtin (this
//   toolchain's clang-23 form), hardware row padding reproducing the
//   bank-conflict-free LDS layout; async global->LDS copies as fallback.
// ---------------------------------------------------------------------------

typedef __attribute__((ext_vector_type(16))) __bf16 v16bf;
typedef __attribute__((ext_vector_type(8)))  float  v8f;
typedef __attribute__((ext_vector_type(4)))  unsigned int u32x4;
typedef __attribute__((ext_vector_type(8)))  int i32x8;
typedef __attribute__((ext_vector_type(4)))  int i32x4;

#define DEPTH   4
#define NEMB    8192
#define EDIM    256
#define NROW    8192            // B*T = 8*1024
#define NSEG    8               // K split into 8 segments (partial argmins)
#define SEG     (NEMB / NSEG)   // 1024 codes per segment
#define CHUNK   64              // codes staged in LDS per iteration
#define NCHUNK  (SEG / CHUNK)   // 16
#define LDSROW  264             // padded LDS row stride (halves): kills bank conflicts

#define OUT_ELEMS  ((size_t)NROW * EDIM)   // 2097152
#define LOSS_OFF   OUT_ELEMS
#define CODES_OFF  (OUT_ELEMS + 1)

#if __has_builtin(__builtin_amdgcn_tensor_load_to_lds) && __has_builtin(__builtin_amdgcn_s_wait_tensorcnt)
#define USE_TDM 1
#else
#define USE_TDM 0
#endif

union FragU { v16bf v; uint32_t u[8]; };

static __device__ __forceinline__ uint16_t f32_to_bf16(float f) {
    uint32_t u = __float_as_uint(f);
    u += 0x7fffu + ((u >> 16) & 1u);     // round-to-nearest-even
    return (uint16_t)(u >> 16);
}

// Async global->LDS 16B copy (CDNA5). VDST = VGPR with LDS byte offset
// (low 32 bits of a generic LDS pointer), VADDR = 64-bit global address.
static __device__ __forceinline__ void async_g2l_b128(uint32_t lds_off, const void* gaddr) {
    asm volatile("global_load_async_to_lds_b128 %0, %1, off"
                 :: "v"(lds_off), "v"((uint64_t)(size_t)gaddr) : "memory");
}
static __device__ __forceinline__ void wait_asynccnt0() {
    asm volatile("s_wait_asynccnt 0x0" ::: "memory");
}

#if USE_TDM
// One TDM descriptor moves a 64x256 bf16 tile (rows of 512B) into LDS,
// inserting 16B of padding after every row (pad_interval=6 -> every 64x8B,
// pad_amount=3 -> 4 dwords) to reproduce the LDSROW=264-half layout.
static __device__ __forceinline__ void tdm_load_tile(uint32_t lds_off, const void* gsrc) {
    const uint64_t ga = (uint64_t)(size_t)gsrc;
    u32x4 g0;
    g0[0] = 1u;                                                  // count=1, user mode
    g0[1] = lds_off;                                             // LDS byte address
    g0[2] = (uint32_t)ga;                                        // global addr lo
    g0[3] = (uint32_t)((ga >> 32) & 0x1FFFFFFu) | (2u << 30);    // global addr hi, type=2
    i32x8 g1;
    g1[0] = (int)((1u << 16)     // data_size = 2 bytes
                | (1u << 20)     // pad_enable
                | (6u << 22)     // pad_interval: every 64 x 8B = 512B (one row)
                | (3u << 25));   // pad_amount: 4 dwords = 16B
    g1[1] = (int)(((uint32_t)EDIM & 0xFFFFu) << 16);             // tensor_dim0 = 256 (lo16)
    g1[2] = (int)(((uint32_t)CHUNK & 0xFFFFu) << 16);            // dim0 hi=0, tensor_dim1 = 64 (lo16)
    g1[3] = (int)((uint32_t)EDIM << 16);                         // dim1 hi=0, tile_dim0 = 256
    g1[4] = (int)(uint32_t)CHUNK;                                // tile_dim1 = 64, tile_dim2 = 0
    g1[5] = (int)(uint32_t)EDIM;                                 // tensor_dim0_stride = 256
    g1[6] = 0;
    g1[7] = 0;
    const i32x4 z4 = {0, 0, 0, 0};
    const i32x8 z8 = {0, 0, 0, 0, 0, 0, 0, 0};
    __builtin_amdgcn_tensor_load_to_lds(g0, g1, z4, z4, z8, 0);
}
#endif

// ---------------------------------------------------------------------------
// Prep 1: codebooks f32 -> bf16, plus ||c||^2 per code row. One wave per row.
// ---------------------------------------------------------------------------
__global__ void rq_cb_prep(const float* __restrict__ cb,
                           uint16_t* __restrict__ cb16,
                           float* __restrict__ cnorm) {
    const int wave = threadIdx.x >> 5, lane = threadIdx.x & 31;
    const int row  = blockIdx.x * 8 + wave;            // 0 .. DEPTH*NEMB-1
    const float* src = cb + (size_t)row * EDIM + lane * 8;
    float4 a = ((const float4*)src)[0];
    float4 b = ((const float4*)src)[1];
    float ss = a.x*a.x + a.y*a.y + a.z*a.z + a.w*a.w
             + b.x*b.x + b.y*b.y + b.z*b.z + b.w*b.w;
    uint4 p;
    p.x = (uint32_t)f32_to_bf16(a.x) | ((uint32_t)f32_to_bf16(a.y) << 16);
    p.y = (uint32_t)f32_to_bf16(a.z) | ((uint32_t)f32_to_bf16(a.w) << 16);
    p.z = (uint32_t)f32_to_bf16(b.x) | ((uint32_t)f32_to_bf16(b.y) << 16);
    p.w = (uint32_t)f32_to_bf16(b.z) | ((uint32_t)f32_to_bf16(b.w) << 16);
    *(uint4*)(cb16 + (size_t)row * EDIM + lane * 8) = p;
    #pragma unroll
    for (int m = 16; m >= 1; m >>= 1) ss += __shfl_xor(ss, m, 32);
    if (lane == 0) cnorm[row] = ss;
}

// ---------------------------------------------------------------------------
// Prep 2: residual_0 = x (f32 copy + bf16 mirror), zero loss accumulators.
// ---------------------------------------------------------------------------
__global__ void rq_res_init(const float* __restrict__ x,
                            float* __restrict__ resf,
                            uint16_t* __restrict__ res16,
                            float* __restrict__ lsum) {
    const size_t i = ((size_t)blockIdx.x * 256 + threadIdx.x) * 4;
    float4 v = *(const float4*)(x + i);
    *(float4*)(resf + i) = v;
    uint2 p;
    p.x = (uint32_t)f32_to_bf16(v.x) | ((uint32_t)f32_to_bf16(v.y) << 16);
    p.y = (uint32_t)f32_to_bf16(v.z) | ((uint32_t)f32_to_bf16(v.w) << 16);
    *(uint2*)(res16 + i) = p;
    if (blockIdx.x == 0 && threadIdx.x < DEPTH) lsum[threadIdx.x] = 0.0f;
}

// ---------------------------------------------------------------------------
// Fused distance GEMM + argmin. 4 waves/WG; each wave owns 16 rows; WG shares
// a double-buffered 64-code LDS tile streamed by the TDM (or async copies);
// grid = (rows/64, NSEG). Accumulator init = -0.5*||c||^2 so argmin(dist) ==
// argmax(C).
// ---------------------------------------------------------------------------
__global__ __launch_bounds__(128) void rq_gemm_argmin(
        const uint16_t* __restrict__ res16,
        const uint16_t* __restrict__ cb16_l,
        const float*    __restrict__ cnorm_l,
        float* __restrict__ pval, int* __restrict__ pidx) {
    __shared__ uint16_t cbuf[2][CHUNK * LDSROW];   // 2 x 33792 B
    __shared__ float    cnbuf[SEG];                // 4 KB

    const int tid  = threadIdx.x;
    const int lane = tid & 31, wave = tid >> 5;
    const int hi   = lane >> 4, lr = lane & 15;
    const int seg  = blockIdx.y;
    const int row  = blockIdx.x * 64 + wave * 16 + lr;
    const int seg_base = seg * SEG;

    // ---- stage chunk 0 into buffer 0 ----------------------------------
#if USE_TDM
    if (wave == 0) {
        tdm_load_tile((uint32_t)(size_t)&cbuf[0][0],
                      cb16_l + (size_t)seg_base * EDIM);
    }
#else
    {
        const uint16_t* src = cb16_l + (size_t)seg_base * EDIM;
        #pragma unroll
        for (int it = 0; it < (CHUNK * EDIM / 8) / 128; ++it) {   // 16 iters
            int q  = tid + it * 128;
            int r  = q >> 5;
            int c8 = (q & 31) * 8;
            async_g2l_b128((uint32_t)(size_t)&cbuf[0][r * LDSROW + c8],
                           src + (size_t)r * EDIM + c8);
        }
    }
#endif
    // ---- stage this segment's ||c||^2 into LDS ------------------------
    {
        const float4* s = (const float4*)(cnorm_l + seg_base);
        float4*       d = (float4*)cnbuf;
        #pragma unroll
        for (int i = 0; i < SEG / 4 / 128; ++i) d[tid + i * 128] = s[tid + i * 128];
    }

    // ---- B fragments (16 residual rows x 256 dims), reused all segment.
    // 16-bit B 32x16 layout: halves 0..7 = K 8*hi.., halves 8..15 = K 16+8*hi..
    FragU Bf[8];
    {
        const uint16_t* rb = res16 + (size_t)row * EDIM + hi * 8;
        #pragma unroll
        for (int s = 0; s < 8; ++s) {
            const uint4* p = (const uint4*)(rb + s * 32);
            uint4 q0 = p[0], q1 = p[2];
            Bf[s].u[0] = q0.x; Bf[s].u[1] = q0.y; Bf[s].u[2] = q0.z; Bf[s].u[3] = q0.w;
            Bf[s].u[4] = q1.x; Bf[s].u[5] = q1.y; Bf[s].u[6] = q1.z; Bf[s].u[7] = q1.w;
        }
    }

#if USE_TDM
    __builtin_amdgcn_s_wait_tensorcnt(0);   // no-op for waves 1..3
#else
    wait_asynccnt0();
#endif
    __syncthreads();

    float best = -3.4e38f;
    int   bidx = 0;

    for (int ch = 0; ch < NCHUNK; ++ch) {
        const int cur = ch & 1;
        // ---- prefetch next chunk into the other buffer (overlaps wmmas)
        if (ch + 1 < NCHUNK) {
#if USE_TDM
            if (wave == 0) {
                tdm_load_tile((uint32_t)(size_t)&cbuf[cur ^ 1][0],
                              cb16_l + (size_t)(seg_base + (ch + 1) * CHUNK) * EDIM);
            }
#else
            const uint16_t* src = cb16_l + (size_t)(seg_base + (ch + 1) * CHUNK) * EDIM;
            uint16_t* dstbuf = &cbuf[cur ^ 1][0];
            #pragma unroll
            for (int it = 0; it < (CHUNK * EDIM / 8) / 128; ++it) {
                int q  = tid + it * 128;
                int r  = q >> 5;
                int c8 = (q & 31) * 8;
                async_g2l_b128((uint32_t)(size_t)(dstbuf + r * LDSROW + c8),
                               src + (size_t)r * EDIM + c8);
            }
#endif
        }

        // ---- compute on current buffer --------------------------------
        const uint16_t* base = &cbuf[cur][0];
        const int code0 = seg_base + ch * CHUNK;
        #pragma unroll
        for (int sc = 0; sc < CHUNK / 16; ++sc) {
            // Accumulator init: lane holds codes {code0+sc*16+8*hi+j}, j=0..7
            const float* cn = cnbuf + ch * CHUNK + sc * 16 + hi * 8;
            float4 c0 = ((const float4*)cn)[0];
            float4 c1 = ((const float4*)cn)[1];
            v8f C;
            C[0] = -0.5f * c0.x; C[1] = -0.5f * c0.y;
            C[2] = -0.5f * c0.z; C[3] = -0.5f * c0.w;
            C[4] = -0.5f * c1.x; C[5] = -0.5f * c1.y;
            C[6] = -0.5f * c1.z; C[7] = -0.5f * c1.w;

            // 16-bit A 16x32 layout: lane<16 -> halves [k0..k0+7],[k0+16..23]
            //                        lane>=16 -> halves [k0+8..15],[k0+24..31]
            const uint16_t* arow = base + (sc * 16 + lr) * LDSROW + hi * 8;
            FragU Afr[2];
            {   // preload step 0
                const uint4 q0 = *(const uint4*)(arow);
                const uint4 q1 = *(const uint4*)(arow + 16);
                Afr[0].u[0] = q0.x; Afr[0].u[1] = q0.y; Afr[0].u[2] = q0.z; Afr[0].u[3] = q0.w;
                Afr[0].u[4] = q1.x; Afr[0].u[5] = q1.y; Afr[0].u[6] = q1.z; Afr[0].u[7] = q1.w;
            }
            #pragma unroll
            for (int s = 0; s < 8; ++s) {
                if (s + 1 < 8) {   // software-pipeline next A fragment
                    FragU& An = Afr[(s + 1) & 1];
                    const uint4 q0 = *(const uint4*)(arow + (s + 1) * 32);
                    const uint4 q1 = *(const uint4*)(arow + (s + 1) * 32 + 16);
                    An.u[0] = q0.x; An.u[1] = q0.y; An.u[2] = q0.z; An.u[3] = q0.w;
                    An.u[4] = q1.x; An.u[5] = q1.y; An.u[6] = q1.z; An.u[7] = q1.w;
                }
                C = __builtin_amdgcn_wmma_f32_16x16x32_bf16(
                        false, Afr[s & 1].v, false, Bf[s].v, (short)0, C, false, false);
            }
            // Running argmax (ascending code order + strict '>' keeps the
            // first-minimal index, matching jnp.argmin).
            #pragma unroll
            for (int j = 0; j < 8; ++j) {
                float v  = C[j];
                int   ci = code0 + sc * 16 + hi * 8 + j;
                if (v > best) { best = v; bidx = ci; }
            }
        }

#if USE_TDM
        __builtin_amdgcn_s_wait_tensorcnt(0);   // next tile resident (wave 0)
#else
        wait_asynccnt0();
#endif
        __syncthreads();      // publish tile to all waves
    }

    // Combine the two half-waves tracking the same row (tie -> smaller idx).
    float ob = __shfl_xor(best, 16, 32);
    int   oi = __shfl_xor(bidx, 16, 32);
    if (ob > best || (ob == best && oi < bidx)) { best = ob; bidx = oi; }
    if (hi == 0) {
        pval[(size_t)row * NSEG + seg] = best;
        pidx[(size_t)row * NSEG + seg] = bidx;
    }
}

// ---------------------------------------------------------------------------
// Finalize a level: reduce NSEG partials -> code, gather f32 codebook row,
// residual -= quant (f32 + bf16 mirror), codes out, loss accumulation.
// losses[i] == mean(residual_after_i ^2); out == x - residual_final.
// One wave per row.
// ---------------------------------------------------------------------------
__global__ void rq_finalize(const float* __restrict__ pval,
                            const int*   __restrict__ pidx,
                            const float* __restrict__ cb_l,
                            float* __restrict__ resf,
                            uint16_t* __restrict__ res16,
                            const float* __restrict__ x,
                            float* __restrict__ dout,
                            float* __restrict__ lsum,
                            int level) {
    const int wave = threadIdx.x >> 5, lane = threadIdx.x & 31;
    const int row  = blockIdx.x * 8 + wave;

    float pv = pval[(size_t)row * NSEG + (lane & 7)];
    int   pi = pidx[(size_t)row * NSEG + (lane & 7)];
    #pragma unroll
    for (int m = 1; m <= 4; m <<= 1) {
        float ov = __shfl_xor(pv, m, 32);
        int   oi = __shfl_xor(pi, m, 32);
        if (ov > pv || (ov == pv && oi < pi)) { pv = ov; pi = oi; }
    }

    const float* q = cb_l + (size_t)pi * EDIM + lane * 8;
    float*       r = resf + (size_t)row * EDIM + lane * 8;
    float4 q0 = ((const float4*)q)[0], q1 = ((const float4*)q)[1];
    float4 r0 = ((const float4*)r)[0], r1 = ((const float4*)r)[1];
    r0.x -= q0.x; r0.y -= q0.y; r0.z -= q0.z; r0.w -= q0.w;
    r1.x -= q1.x; r1.y -= q1.y; r1.z -= q1.z; r1.w -= q1.w;
    ((float4*)r)[0] = r0; ((float4*)r)[1] = r1;

    uint4 p;
    p.x = (uint32_t)f32_to_bf16(r0.x) | ((uint32_t)f32_to_bf16(r0.y) << 16);
    p.y = (uint32_t)f32_to_bf16(r0.z) | ((uint32_t)f32_to_bf16(r0.w) << 16);
    p.z = (uint32_t)f32_to_bf16(r1.x) | ((uint32_t)f32_to_bf16(r1.y) << 16);
    p.w = (uint32_t)f32_to_bf16(r1.z) | ((uint32_t)f32_to_bf16(r1.w) << 16);
    *(uint4*)(res16 + (size_t)row * EDIM + lane * 8) = p;

    float ss = r0.x*r0.x + r0.y*r0.y + r0.z*r0.z + r0.w*r0.w
             + r1.x*r1.x + r1.y*r1.y + r1.z*r1.z + r1.w*r1.w;
    #pragma unroll
    for (int m = 16; m >= 1; m >>= 1) ss += __shfl_xor(ss, m, 32);
    if (lane == 0) {
        atomicAdd(lsum + level, ss);
        dout[CODES_OFF + (size_t)row * DEPTH + level] = (float)pi;
    }

    if (level == DEPTH - 1) {
        const float* xr = x + (size_t)row * EDIM + lane * 8;
        float4 x0 = ((const float4*)xr)[0], x1 = ((const float4*)xr)[1];
        float4 o0, o1;
        o0.x = x0.x - r0.x; o0.y = x0.y - r0.y; o0.z = x0.z - r0.z; o0.w = x0.w - r0.w;
        o1.x = x1.x - r1.x; o1.y = x1.y - r1.y; o1.z = x1.z - r1.z; o1.w = x1.w - r1.w;
        float* od = dout + (size_t)row * EDIM + lane * 8;
        ((float4*)od)[0] = o0; ((float4*)od)[1] = o1;
    }
}

__global__ void rq_loss(const float* __restrict__ lsum, float* __restrict__ dout) {
    if (threadIdx.x == 0 && blockIdx.x == 0) {
        float s = lsum[0] + lsum[1] + lsum[2] + lsum[3];
        dout[LOSS_OFF] = 0.25f * s / ((float)DEPTH * (float)NROW * (float)EDIM);
    }
}

// ---------------------------------------------------------------------------
extern "C" void kernel_launch(void* const* d_in, const int* in_sizes, int n_in,
                              void* d_out, int out_size, void* d_ws, size_t ws_size,
                              hipStream_t stream) {
    const float* x  = (const float*)d_in[0];   // [8,1024,256] f32
    const float* cb = (const float*)d_in[1];   // [4,8192,256] f32
    float* out = (float*)d_out;

    char* ws = (char*)d_ws;
    size_t o = 0;
    uint16_t* cb16  = (uint16_t*)(ws + o); o += (size_t)DEPTH * NEMB * EDIM * 2; // 16 MB
    float*    cnorm = (float*)(ws + o);    o += (size_t)DEPTH * NEMB * 4;        // 128 KB
    float*    resf  = (float*)(ws + o);    o += (size_t)NROW * EDIM * 4;         // 8 MB
    uint16_t* res16 = (uint16_t*)(ws + o); o += (size_t)NROW * EDIM * 2;         // 4 MB
    float*    pval  = (float*)(ws + o);    o += (size_t)NROW * NSEG * 4;         // 256 KB
    int*      pidx  = (int*)(ws + o);      o += (size_t)NROW * NSEG * 4;         // 256 KB
    float*    lsum  = (float*)(ws + o);    o += 256;

    rq_cb_prep<<<DEPTH * NEMB / 8, 256, 0, stream>>>(cb, cb16, cnorm);
    rq_res_init<<<(NROW * EDIM / 4) / 256, 256, 0, stream>>>(x, resf, res16, lsum);

    for (int l = 0; l < DEPTH; ++l) {
        rq_gemm_argmin<<<dim3(NROW / 64, NSEG), 128, 0, stream>>>(
            res16, cb16 + (size_t)l * NEMB * EDIM, cnorm + (size_t)l * NEMB,
            pval, pidx);
        rq_finalize<<<NROW / 8, 256, 0, stream>>>(
            pval, pidx, cb + (size_t)l * NEMB * EDIM,
            resf, res16, x, out, lsum, l);
    }
    rq_loss<<<1, 1, 0, stream>>>(lsum, out);
}
